// OptimizedTripletAttention_78958678769828
// MI455X (gfx1250) — compile-verified, matrix operand order checked by reading, not run on hardware
//
#include <hip/hip_runtime.h>

// ---------------------------------------------------------------------------
// Triplet attention for MI455X (gfx1250, wave32).
// All matmuls via v_wmma_f32_16x16x32_bf16 (fp32 accumulate).
// GEMMs use the Tensor Data Mover (tensor_load_to_lds) with double-buffered
// LDS so DMA of K-slab k+32 overlaps WMMA on slab k.
// ---------------------------------------------------------------------------

typedef __bf16 bf16;
typedef __bf16        v16bf __attribute__((ext_vector_type(16)));
typedef float         v8f   __attribute__((ext_vector_type(8)));
typedef unsigned int  v4ui  __attribute__((ext_vector_type(4)));
typedef int           v4i   __attribute__((ext_vector_type(4)));
typedef int           v8i   __attribute__((ext_vector_type(8)));

#define BB   2
#define TT   2048
#define DD   1024
#define HH   16
#define HDD  64
#define SCL  0.125f   // 1/sqrt(64)

#if defined(__has_builtin)
#if __has_builtin(__builtin_amdgcn_tensor_load_to_lds) && \
    __has_builtin(__builtin_amdgcn_s_wait_tensorcnt)
#define HAVE_TDM 1
#endif
#endif
#ifndef HAVE_TDM
#define HAVE_TDM 0
#endif

static __device__ __forceinline__ v8f wmma_bf16(v16bf a, v16bf b, v8f c) {
  return __builtin_amdgcn_wmma_f32_16x16x32_bf16(false, a, false, b, (short)0, c,
                                                 false, false);
}

#if HAVE_TDM
// LDS byte offset of a __shared__ object (aperture low bits == LDS offset).
static __device__ __forceinline__ unsigned lds_off(const void* p) {
  return (unsigned)(unsigned long long)(size_t)p;
}

// Issue one TDM 2-D tile load: global (row-major, stride0 elements per row,
// bf16) -> LDS (tile rows contiguous). D# per CDNA5 ISA ch.8.
static __device__ __forceinline__ void tdm_load_2d(unsigned lds_addr,
                                                   const void* gptr,
                                                   unsigned tensor_d0,
                                                   unsigned tensor_d1,
                                                   unsigned tile_d0,
                                                   unsigned tile_d1,
                                                   unsigned stride0) {
  unsigned long long ga = (unsigned long long)(size_t)gptr;
  v4ui g0;
  g0[0] = 1u;                                   // count=1 (valid), user mode
  g0[1] = lds_addr;                             // lds_addr (bytes)
  g0[2] = (unsigned)(ga & 0xFFFFFFFFu);         // global_addr[31:0]
  g0[3] = (unsigned)((ga >> 32) & 0x01FFFFFFu)  // global_addr[56:32]
          | (2u << 30);                         // type = 2 ("image")
  v8i g1;
  g1[0] = (int)(1u << 16);                      // data_size=1 -> 2 bytes
  g1[1] = (int)((tensor_d0 & 0xFFFFu) << 16);   // tensor_dim0[15:0]
  g1[2] = (int)(((tensor_d0 >> 16) & 0xFFFFu) |
                ((tensor_d1 & 0xFFFFu) << 16)); // dim0[31:16] | dim1[15:0]
  g1[3] = (int)(((tensor_d1 >> 16) & 0xFFFFu) |
                ((tile_d0 & 0xFFFFu) << 16));   // dim1[31:16] | tile_dim0
  g1[4] = (int)(tile_d1 & 0xFFFFu);             // tile_dim1 (tile_dim2=0)
  g1[5] = (int)stride0;                         // tensor_dim0_stride[31:0]
  g1[6] = 0;                                    // stride hi | dim1_stride lo
  g1[7] = 0;
  v4i z4 = {0, 0, 0, 0};
#if defined(__clang_major__) && (__clang_major__ >= 23)
  v8i z8 = {0, 0, 0, 0, 0, 0, 0, 0};
  __builtin_amdgcn_tensor_load_to_lds(g0, g1, z4, z4, z8, 0);
#else
  __builtin_amdgcn_tensor_load_to_lds(g0, g1, z4, z4, 0);
#endif
}
#endif  // HAVE_TDM

// ---------------------------------------------------------------------------
// fp32 -> bf16 convert (straight)
// ---------------------------------------------------------------------------
__global__ void __launch_bounds__(256)
cvt_bf16_kernel(const float* __restrict__ in, bf16* __restrict__ out, int n) {
  int i = blockIdx.x * blockDim.x + threadIdx.x;
  if (i < n) out[i] = (bf16)in[i];
}

// fp32 [K,N] row-major -> bf16 [N,K] row-major (transpose)
__global__ void __launch_bounds__(256)
cvtT_bf16_kernel(const float* __restrict__ in, bf16* __restrict__ out,
                 int K, int N) {
  int i = blockIdx.x * blockDim.x + threadIdx.x;
  if (i < K * N) {
    int k = i / N, n = i - k * N;
    out[(size_t)n * K + k] = (bf16)in[i];
  }
}

// ---------------------------------------------------------------------------
// Shared GEMM mainloop: C(64x128 tile) += A(64xK) @ Bt(128xK)^T, bf16 WMMA.
// TDM path: double-buffered LDS; wave 0 issues DMA for slab k+32 while all
// waves run WMMA on slab k. Fallback: manual staged loads + prefetch.
// ---------------------------------------------------------------------------
#define GEMM_MAINLOOP(A_, Bt_, K_)                                             \
  __shared__ __align__(128) bf16 As[2][64][32];                                \
  __shared__ __align__(128) bf16 Bs[2][128][32];                               \
  const int tid = threadIdx.x;                                                 \
  const int lane = tid & 31, w = tid >> 5;                                     \
  const int wr = w & 3, wc = w >> 2;                                           \
  const int lr = lane & 15, lh = lane >> 4;                                    \
  const int m0 = blockIdx.y * 64, n0 = blockIdx.x * 128;                       \
  v8f acc[4];                                                                  \
  _Pragma("unroll") for (int i = 0; i < 4; ++i)                                \
      _Pragma("unroll") for (int e = 0; e < 8; ++e) acc[i][e] = 0.0f;          \
  GEMM_STAGE_AND_COMPUTE(A_, Bt_, K_)

#if HAVE_TDM
#define GEMM_STAGE_AND_COMPUTE(A_, Bt_, K_)                                    \
  {                                                                            \
    const unsigned ldsA0 = lds_off(&As[0][0][0]);                              \
    const unsigned ldsA1 = lds_off(&As[1][0][0]);                              \
    const unsigned ldsB0 = lds_off(&Bs[0][0][0]);                              \
    const unsigned ldsB1 = lds_off(&Bs[1][0][0]);                              \
    if (w == 0) { /* prologue DMA for k0 = 0 into buffer 0 */                  \
      tdm_load_2d(ldsA0, (A_) + (size_t)m0 * (K_), (K_), 64, 32, 64, (K_));    \
      tdm_load_2d(ldsB0, (Bt_) + (size_t)n0 * (K_), (K_), 128, 32, 128, (K_)); \
      __builtin_amdgcn_s_wait_tensorcnt((short)0);                             \
    }                                                                          \
    __syncthreads();                                                           \
    for (int k0 = 0; k0 < (K_); k0 += 32) {                                    \
      const int cur = (k0 >> 5) & 1;                                           \
      if (w == 0 && k0 + 32 < (K_)) { /* DMA next slab into other buffer */    \
        tdm_load_2d(cur ? ldsA0 : ldsA1,                                       \
                    (A_) + (size_t)m0 * (K_) + k0 + 32, (K_), 64, 32, 64,      \
                    (K_));                                                     \
        tdm_load_2d(cur ? ldsB0 : ldsB1,                                       \
                    (Bt_) + (size_t)n0 * (K_) + k0 + 32, (K_), 128, 32, 128,   \
                    (K_));                                                     \
      }                                                                        \
      v16bf af = *(const v16bf*)&As[cur][wr * 16 + lr][lh * 16];               \
      v16bf b0 = *(const v16bf*)&Bs[cur][wc * 64 + 0 + lr][lh * 16];           \
      v16bf b1 = *(const v16bf*)&Bs[cur][wc * 64 + 16 + lr][lh * 16];          \
      v16bf b2 = *(const v16bf*)&Bs[cur][wc * 64 + 32 + lr][lh * 16];          \
      v16bf b3 = *(const v16bf*)&Bs[cur][wc * 64 + 48 + lr][lh * 16];          \
      acc[0] = wmma_bf16(af, b0, acc[0]);                                      \
      acc[1] = wmma_bf16(af, b1, acc[1]);                                      \
      acc[2] = wmma_bf16(af, b2, acc[2]);                                      \
      acc[3] = wmma_bf16(af, b3, acc[3]);                                      \
      if (k0 + 32 < (K_)) {                                                    \
        if (w == 0) __builtin_amdgcn_s_wait_tensorcnt((short)0);               \
        __syncthreads();                                                       \
      }                                                                        \
    }                                                                          \
  }
#else
#define GEMM_STAGE_AND_COMPUTE(A_, Bt_, K_)                                    \
  {                                                                            \
    const int arow = tid >> 2, akk = (tid & 3) * 8;                            \
    const int brow = tid >> 1, bkk = (tid & 1) * 16;                           \
    for (int k0 = 0; k0 < (K_); k0 += 32) {                                    \
      __syncthreads();                                                         \
      *(v4ui*)&As[0][arow][akk] =                                              \
          *(const v4ui*)((A_) + (size_t)(m0 + arow) * (K_) + k0 + akk);        \
      *(v16bf*)&Bs[0][brow][bkk] =                                             \
          *(const v16bf*)((Bt_) + (size_t)(n0 + brow) * (K_) + k0 + bkk);      \
      if (k0 + 32 < (K_)) {                                                    \
        __builtin_prefetch((A_) + (size_t)(m0 + arow) * (K_) + k0 + 32 + akk,  \
                           0, 3);                                              \
        __builtin_prefetch((Bt_) + (size_t)(n0 + brow) * (K_) + k0 + 32 + bkk, \
                           0, 3);                                              \
      }                                                                        \
      __syncthreads();                                                         \
      v16bf af = *(const v16bf*)&As[0][wr * 16 + lr][lh * 16];                 \
      v16bf b0 = *(const v16bf*)&Bs[0][wc * 64 + 0 + lr][lh * 16];             \
      v16bf b1 = *(const v16bf*)&Bs[0][wc * 64 + 16 + lr][lh * 16];            \
      v16bf b2 = *(const v16bf*)&Bs[0][wc * 64 + 32 + lr][lh * 16];            \
      v16bf b3 = *(const v16bf*)&Bs[0][wc * 64 + 48 + lr][lh * 16];            \
      acc[0] = wmma_bf16(af, b0, acc[0]);                                      \
      acc[1] = wmma_bf16(af, b1, acc[1]);                                      \
      acc[2] = wmma_bf16(af, b2, acc[2]);                                      \
      acc[3] = wmma_bf16(af, b3, acc[3]);                                      \
    }                                                                          \
  }
#endif

// ---------------------------------------------------------------------------
// GEMM 1: qkvt = x @ Wqkvt + b   (M=4096, N=4096, K=1024)
// Epilogue scatters into q/k/v/t buffers laid out [B,H,T,HD] (bf16).
// ---------------------------------------------------------------------------
__global__ void __launch_bounds__(256)
gemm_qkvt_kernel(const bf16* __restrict__ A, const bf16* __restrict__ Bt,
                 const float* __restrict__ bias,
                 bf16* __restrict__ qh, bf16* __restrict__ kh,
                 bf16* __restrict__ vh, bf16* __restrict__ th) {
  GEMM_MAINLOOP(A, Bt, DD)

#pragma unroll
  for (int nt = 0; nt < 4; ++nt) {
    int n = n0 + wc * 64 + nt * 16 + lr;
    int which = n >> 10;            // 0=q 1=k 2=v 3=t
    int h = (n >> 6) & (HH - 1);
    int hd = n & (HDD - 1);
    bf16* dst = (which == 0) ? qh : (which == 1) ? kh : (which == 2) ? vh : th;
    float bv = bias[n];
#pragma unroll
    for (int r = 0; r < 8; ++r) {
      int m = m0 + wr * 16 + r + 8 * lh;
      int b = m >> 11, t = m & (TT - 1);
      dst[(((size_t)(b * HH + h)) * TT + t) * HDD + hd] = (bf16)(acc[nt][r] + bv);
    }
  }
}

// ---------------------------------------------------------------------------
// GEMM 2: out = Ob @ Wout + bout  (M=4096, N=1024, K=1024), fp32 output.
// ---------------------------------------------------------------------------
__global__ void __launch_bounds__(256)
gemm_out_kernel(const bf16* __restrict__ A, const bf16* __restrict__ Bt,
                const float* __restrict__ bias, float* __restrict__ out) {
  GEMM_MAINLOOP(A, Bt, DD)

#pragma unroll
  for (int nt = 0; nt < 4; ++nt) {
    int n = n0 + wc * 64 + nt * 16 + lr;
    float bv = bias[n];
#pragma unroll
    for (int r = 0; r < 8; ++r) {
      int m = m0 + wr * 16 + r + 8 * lh;
      out[(size_t)m * DD + n] = acc[nt][r] + bv;
    }
  }
}

// ---------------------------------------------------------------------------
// Elementwise: qt = q * t  (bf16 in/out, fp32 math)
// ---------------------------------------------------------------------------
__global__ void __launch_bounds__(256)
qt_kernel(const bf16* __restrict__ q, const bf16* __restrict__ t,
          bf16* __restrict__ qt, int n) {
  int i = blockIdx.x * blockDim.x + threadIdx.x;
  if (i < n) qt[i] = (bf16)((float)q[i] * (float)t[i]);
}

// ---------------------------------------------------------------------------
// Causal prefix scan: kvc[b,h,t,:] = cumsum_t(k*v*SCALE); also emit vT[b,h,hd,t]
// ---------------------------------------------------------------------------
__global__ void __launch_bounds__(256)
scan_kernel(const bf16* __restrict__ kh, const bf16* __restrict__ vh,
            bf16* __restrict__ kvc, bf16* __restrict__ vT) {
  int tid = blockIdx.x * blockDim.x + threadIdx.x;
  if (tid >= BB * HH * HDD) return;
  int bh = tid / HDD, hd = tid - bh * HDD;
  float acc = 0.0f;
  for (int t = 0; t < TT; ++t) {
    size_t idx = ((size_t)bh * TT + t) * HDD + hd;
    acc += (float)kh[idx] * (float)vh[idx] * SCL;
    kvc[idx] = (bf16)acc;
    vT[((size_t)bh * HDD + hd) * TT + t] = vh[idx];
  }
}

// ---------------------------------------------------------------------------
// Flash triplet attention.
// grid = (T/64, B*H), block = 128 (4 waves). Wave w owns 16 query rows.
// scores = (q@k^T + (q*t)@kvc^T) * SCALE, causal mask, online softmax, P@V.
// ---------------------------------------------------------------------------
__global__ void __launch_bounds__(128)
attn_kernel(const bf16* __restrict__ qh, const bf16* __restrict__ qth,
            const bf16* __restrict__ kh, const bf16* __restrict__ kvc,
            const bf16* __restrict__ vT, bf16* __restrict__ Ob) {
  __shared__ __align__(64) bf16 Ps[4][16][64];

  const int tid = threadIdx.x;
  const int lane = tid & 31, w = tid >> 5;
  const int lr = lane & 15, lh = lane >> 4;
  const int bh = blockIdx.y;                // b*H + h
  const int qt0 = blockIdx.x * 64 + w * 16; // wave's first query row

  // A-fragments for q and q*t (HD=64 -> 2 K-steps of 32)
  const int qrow = qt0 + lr;
  v16bf aq[2], aqt[2];
#pragma unroll
  for (int s = 0; s < 2; ++s) {
    size_t off = ((size_t)bh * TT + qrow) * HDD + s * 32 + lh * 16;
    aq[s]  = *(const v16bf*)(qh + off);
    aqt[s] = *(const v16bf*)(qth + off);
  }

  v8f oacc[4];
  float mrun[8], lrun[8];
#pragma unroll
  for (int j = 0; j < 4; ++j)
#pragma unroll
    for (int e = 0; e < 8; ++e) oacc[j][e] = 0.0f;
#pragma unroll
  for (int r = 0; r < 8; ++r) { mrun[r] = -1.0e30f; lrun[r] = 0.0f; }

  const int lastkt = (int)blockIdx.x;
  for (int kt = 0; kt <= lastkt; ++kt) {
    // ---- scores tile: 16 x 64 (batched loads, back-to-back WMMA) -------
    v8f sacc[4];
#pragma unroll
    for (int nt = 0; nt < 4; ++nt) {
#pragma unroll
      for (int e = 0; e < 8; ++e) sacc[nt][e] = 0.0f;
      int key = kt * 64 + nt * 16 + lr;
      const bf16* kp = kh + ((size_t)bh * TT + key) * HDD + lh * 16;
      const bf16* cp = kvc + ((size_t)bh * TT + key) * HDD + lh * 16;
      v16bf bk0 = *(const v16bf*)(kp);
      v16bf bk1 = *(const v16bf*)(kp + 32);
      v16bf bc0 = *(const v16bf*)(cp);
      v16bf bc1 = *(const v16bf*)(cp + 32);
      sacc[nt] = wmma_bf16(aq[0], bk0, sacc[nt]);    // dot term
      sacc[nt] = wmma_bf16(aq[1], bk1, sacc[nt]);
      sacc[nt] = wmma_bf16(aqt[0], bc0, sacc[nt]);   // triplet term
      sacc[nt] = wmma_bf16(aqt[1], bc1, sacc[nt]);
#pragma unroll
      for (int e = 0; e < 8; ++e) sacc[nt][e] *= SCL;
    }

    // ---- causal mask on diagonal tile ---------------------------------
    if (kt == lastkt) {
#pragma unroll
      for (int nt = 0; nt < 4; ++nt) {
        int col = kt * 64 + nt * 16 + lr;
#pragma unroll
        for (int r = 0; r < 8; ++r) {
          int row = qt0 + r + 8 * lh;
          if (col > row) sacc[nt][r] = -1.0e30f;
        }
      }
    }

    // ---- online softmax (row stats live in 16-lane halves) -------------
#pragma unroll
    for (int r = 0; r < 8; ++r) {
      float v = sacc[0][r];
#pragma unroll
      for (int nt = 1; nt < 4; ++nt) v = fmaxf(v, sacc[nt][r]);
#pragma unroll
      for (int mk = 1; mk < 16; mk <<= 1) v = fmaxf(v, __shfl_xor(v, mk, 32));
      float mnew = fmaxf(mrun[r], v);
      float alpha = __expf(mrun[r] - mnew);
      mrun[r] = mnew;
      float rsum = 0.0f;
#pragma unroll
      for (int nt = 0; nt < 4; ++nt) {
        sacc[nt][r] = __expf(sacc[nt][r] - mnew);
        rsum += sacc[nt][r];
      }
#pragma unroll
      for (int mk = 1; mk < 16; mk <<= 1) rsum += __shfl_xor(rsum, mk, 32);
      lrun[r] = lrun[r] * alpha + rsum;
#pragma unroll
      for (int j = 0; j < 4; ++j) oacc[j][r] *= alpha;
    }

    // ---- P: C-layout -> LDS -> A-fragments (same-wave, in-order LDS) ---
#pragma unroll
    for (int nt = 0; nt < 4; ++nt)
#pragma unroll
      for (int r = 0; r < 8; ++r)
        Ps[w][r + 8 * lh][nt * 16 + lr] = (bf16)sacc[nt][r];

    v16bf pa[2];
#pragma unroll
    for (int s = 0; s < 2; ++s)
      pa[s] = *(const v16bf*)&Ps[w][lr][s * 32 + lh * 16];

    // ---- O += P @ V  (V^T layout -> contiguous K per lane) --------------
#pragma unroll
    for (int j = 0; j < 4; ++j) {
      int hd = j * 16 + lr;
      const bf16* vp = vT + ((size_t)bh * HDD + hd) * TT + kt * 64 + lh * 16;
      v16bf bv0 = *(const v16bf*)(vp);
      v16bf bv1 = *(const v16bf*)(vp + 32);
      oacc[j] = wmma_bf16(pa[0], bv0, oacc[j]);
      oacc[j] = wmma_bf16(pa[1], bv1, oacc[j]);
    }
  }

  // ---- normalize, write O as [B,T,H,HD] (= [B,T,D]) bf16 ----------------
  const int b = bh >> 4, h = bh & (HH - 1);
#pragma unroll
  for (int r = 0; r < 8; ++r) {
    float inv = 1.0f / lrun[r];
    int row = qt0 + r + 8 * lh;
#pragma unroll
    for (int j = 0; j < 4; ++j) {
      int hd = j * 16 + lr;
      Ob[(((size_t)(b * TT + row)) * HH + h) * HDD + hd] =
          (bf16)(oacc[j][r] * inv);
    }
  }
}

// ---------------------------------------------------------------------------
// Host-side launch
// ---------------------------------------------------------------------------
extern "C" void kernel_launch(void* const* d_in, const int* in_sizes, int n_in,
                              void* d_out, int out_size, void* d_ws,
                              size_t ws_size, hipStream_t stream) {
  (void)in_sizes; (void)n_in; (void)out_size; (void)ws_size;

  const float* x     = (const float*)d_in[0];   // [B,T,D]
  const float* Wqkvt = (const float*)d_in[1];   // [D,4D]
  const float* bqkvt = (const float*)d_in[2];   // [4D]
  const float* Wout  = (const float*)d_in[3];   // [D,D]
  const float* bout  = (const float*)d_in[4];   // [D]
  float* out = (float*)d_out;                   // [B,T,D]

  const size_t nBTD  = (size_t)BB * TT * DD;         // 4,194,304
  const size_t nBHTD = (size_t)BB * HH * TT * HDD;   // same count

  char* ws = (char*)d_ws;
  bf16* xb  = (bf16*)ws;                 ws += nBTD * 2;
  bf16* WqT = (bf16*)ws;                 ws += (size_t)4 * DD * DD * 2;
  bf16* WoT = (bf16*)ws;                 ws += (size_t)DD * DD * 2;
  bf16* qh  = (bf16*)ws;                 ws += nBHTD * 2;
  bf16* kh  = (bf16*)ws;                 ws += nBHTD * 2;
  bf16* vh  = (bf16*)ws;                 ws += nBHTD * 2;
  bf16* th  = (bf16*)ws;                 ws += nBHTD * 2;
  bf16* qtb = (bf16*)ws;                 ws += nBHTD * 2;
  bf16* kvc = (bf16*)ws;                 ws += nBHTD * 2;
  bf16* vT  = (bf16*)ws;                 ws += nBHTD * 2;
  bf16* Ob  = (bf16*)ws;                 ws += nBTD * 2;

  // 1) convert inputs to bf16 (weights transposed to [N,K])
  cvt_bf16_kernel<<<(int)((nBTD + 255) / 256), 256, 0, stream>>>(x, xb,
                                                                 (int)nBTD);
  cvtT_bf16_kernel<<<(int)((4u * DD * DD + 255) / 256), 256, 0, stream>>>(
      Wqkvt, WqT, DD, 4 * DD);
  cvtT_bf16_kernel<<<(DD * DD + 255) / 256, 256, 0, stream>>>(Wout, WoT, DD, DD);

  // 2) fused QKVT projection (M=4096, N=4096)
  gemm_qkvt_kernel<<<dim3((4 * DD) / 128, (BB * TT) / 64), 256, 0, stream>>>(
      xb, WqT, bqkvt, qh, kh, vh, th);

  // 3) elementwise q*t ; 4) causal scan of k*v*SCALE + V^T
  qt_kernel<<<(int)((nBHTD + 255) / 256), 256, 0, stream>>>(qh, th, qtb,
                                                            (int)nBHTD);
  scan_kernel<<<(BB * HH * HDD + 255) / 256, 256, 0, stream>>>(kh, vh, kvc, vT);

  // 5) flash triplet attention
  attn_kernel<<<dim3(TT / 64, BB * HH), 128, 0, stream>>>(qh, qtb, kh, kvc, vT,
                                                          Ob);

  // 6) output projection (M=4096, N=1024)
  gemm_out_kernel<<<dim3(DD / 128, (BB * TT) / 64), 256, 0, stream>>>(
      Ob, WoT, bout, out);
}
